// EGNN_var1_12799002542244
// MI455X (gfx1250) — matrix operand order repaired and checked by reading, not compiled, not run on hardware
//
#include <hip/hip_runtime.h>

typedef __attribute__((ext_vector_type(16))) _Float16 v16h;
typedef __attribute__((ext_vector_type(8)))  _Float16 v8h;
typedef __attribute__((ext_vector_type(2)))  _Float16 v2h;
typedef __attribute__((ext_vector_type(8)))  float    v8f;

#define NTOT 32768
#define ETOT 491520
#define FDIM 11
#define HDIM 64
#define GCNT 2048
#define LAY  4
#define RCNT 1024
#define EIN  129
#define NIN  139
#define LSTR 72   // padded LDS row stride (halfs); 72*2=144B keeps 16B alignment

__device__ __forceinline__ float silu_f(float x) { return x / (1.f + __expf(-x)); }

__device__ __forceinline__ v16h frag16(v8h lo, v8h hi) {
  return __builtin_shufflevector(lo, hi, 0, 1, 2, 3, 4, 5, 6, 7,
                                 8, 9, 10, 11, 12, 13, 14, 15);
}

// f32x16 -> f16x16 via packed cvt (v_cvt_pk_rtz_f16_f32): 8 cvts, no pack movs
__device__ __forceinline__ v16h frag16f(v8f lo, v8f hi) {
  union { v16h v; v2h p[8]; } u;
#pragma unroll
  for (int j = 0; j < 4; ++j) {
    u.p[j]     = __builtin_bit_cast(v2h, __builtin_amdgcn_cvt_pkrtz(lo[2 * j], lo[2 * j + 1]));
    u.p[j + 4] = __builtin_bit_cast(v2h, __builtin_amdgcn_cvt_pkrtz(hi[2 * j], hi[2 * j + 1]));
  }
  return u.v;
}

// ---------------------------------------------------------------- small utils
__global__ void k_zero(float* p, int n) {
  int i = blockIdx.x * blockDim.x + threadIdx.x;
  if (i < n) p[i] = 0.f;
}

// Pre-swizzle weight matrix W[K][64] (f32) into WMMA-B fragment-major f16:
// out[((c*4 + t)*32 + lane)*16 + hh] = W[c*32 + (lane>>4)*8 + (hh<8 ? hh : hh+8)][t*16 + (lane&15)]
// (zero-padded past K) so a lane's whole fragment is one contiguous 32B load.
__global__ void k_swz(const float* __restrict__ W, _Float16* __restrict__ out,
                      int K, int nchunks) {
  int i = blockIdx.x * blockDim.x + threadIdx.x;
  if (i >= nchunks * 2048) return;
  int hh   = i & 15;
  int lane = (i >> 4) & 31;
  int t    = (i >> 9) & 3;
  int c    = i >> 11;
  int n    = lane & 15;
  int kk   = c * 32 + ((lane >> 4) << 3) + (hh < 8 ? hh : hh + 8);
  out[i] = (kk < K) ? (_Float16)W[kk * HDIM + t * 16 + n] : (_Float16)0.f;
}

__global__ void k_radial(const float* __restrict__ pos, const int* __restrict__ row,
                         const int* __restrict__ col, float* __restrict__ rad) {
  int e = blockIdx.x * blockDim.x + threadIdx.x;
  if (e >= ETOT) return;
  int r = row[e], c = col[e];
  float dx = pos[r * 3 + 0] - pos[c * 3 + 0];
  float dy = pos[r * 3 + 1] - pos[c * 3 + 1];
  float dz = pos[r * 3 + 2] - pos[c * 3 + 2];
  rad[e] = dx * dx + dy * dy + dz * dz;
}

// h = h0 @ emb_w + emb_b  (K=11); also cache h0 as f16 padded to 16 cols
__global__ void k_embed(const float* __restrict__ h0, const float* __restrict__ w,
                        const float* __restrict__ b, _Float16* __restrict__ h16,
                        _Float16* __restrict__ h016) {
  int i = blockIdx.x * blockDim.x + threadIdx.x;
  if (i >= NTOT * HDIM) return;
  int node = i >> 6, j = i & 63;
  float s = b[j];
#pragma unroll
  for (int k = 0; k < FDIM; ++k) s += h0[node * FDIM + k] * w[k * HDIM + j];
  h16[i] = (_Float16)s;
  if (j < 16)
    h016[node * 16 + j] = (j < FDIM) ? (_Float16)h0[node * FDIM + j] : (_Float16)0.f;
}

// ---------------------------------------------------------------- edge MLP
// One wave = 16 edges. GEMM1: [16x160]x[160x64] (20 WMMA), GEMM2: [16x64]x[64x64]
// (8 WMMA); silu after both, *edge_mask, atomic segment-sum scatter into agg[row].
__global__ void __launch_bounds__(128)
k_edge(const _Float16* __restrict__ h16, const float* __restrict__ rad,
       const int* __restrict__ row, const int* __restrict__ col,
       const float* __restrict__ emask,
       const _Float16* __restrict__ W1s, const float* __restrict__ b1,
       const _Float16* __restrict__ W2s, const float* __restrict__ b2,
       float* __restrict__ agg) {
  __shared__ _Float16 lds[4][16 * LSTR];
  const int wave = threadIdx.x >> 5, lane = threadIdx.x & 31;
  const int tile = blockIdx.x * 4 + wave;
  const int e0 = tile * 16;
  const int m = lane & 15;
  const int kb = (lane >> 4) << 3;    // A-frag K base (0 or 8)
  const int i0 = kb >> 3;             // v8h index of A low half (0 or 1)
  const int n = lane & 15;
  const int rhalf = (lane >> 4) << 3; // C/D row half (0 or 8)
  const v16h* W1f = (const v16h*)W1s;
  const v16h* W2f = (const v16h*)W2s;

  const int eA = e0 + m;
  // prefetch upcoming edge-index stream (speculative; OOB dropped by HW)
  __builtin_prefetch(row + eA + 8192, 0, 3);
  __builtin_prefetch(col + eA + 8192, 0, 3);

  const v8h* hr8 = (const v8h*)(h16 + row[eA] * HDIM);
  const v8h* hc8 = (const v8h*)(h16 + col[eA] * HDIM);

  // A fragments: chunks 0-1 = h[row], 2-3 = h[col], 4 = radial(K=128) + zeros
  v16h a[5];
  a[0] = frag16(hr8[i0],     hr8[i0 + 2]);
  a[1] = frag16(hr8[i0 + 4], hr8[i0 + 6]);
  a[2] = frag16(hc8[i0],     hc8[i0 + 2]);
  a[3] = frag16(hc8[i0 + 4], hc8[i0 + 6]);
  float rv = rad[eA];
  a[4] = v16h{};
  a[4][0] = (lane < 16) ? (_Float16)rv : (_Float16)0.f;

  v8f acc[4] = {};
#pragma unroll
  for (int t = 0; t < 4; ++t) {
#pragma unroll
    for (int c = 0; c < 5; ++c) {
      v16h bf = W1f[(c * 4 + t) * 32 + lane];
      acc[t] = __builtin_amdgcn_wmma_f32_16x16x32_f16(false, a[c], false, bf,
                                                      (short)0, acc[t], false, false);
    }
  }

  // bias + silu, stash tile in LDS to re-shape D-layout -> A-layout
  _Float16* tl = lds[wave];
#pragma unroll
  for (int t = 0; t < 4; ++t) {
    const int cn = t * 16 + n;
    const float bb = b1[cn];
#pragma unroll
    for (int r = 0; r < 8; ++r)
      tl[(rhalf + r) * LSTR + cn] = (_Float16)silu_f(acc[t][r] + bb);
  }
  // wave-private LDS region; LDS ops are in-order within a wave

  const v8h* tl8 = (const v8h*)(tl + m * LSTR);
  v16h a2[2];
  a2[0] = frag16(tl8[i0],     tl8[i0 + 2]);
  a2[1] = frag16(tl8[i0 + 4], tl8[i0 + 6]);

  v8f acc2[4] = {};
#pragma unroll
  for (int t = 0; t < 4; ++t) {
#pragma unroll
    for (int c = 0; c < 2; ++c) {
      v16h bf = W2f[(c * 4 + t) * 32 + lane];
      acc2[t] = __builtin_amdgcn_wmma_f32_16x16x32_f16(false, a2[c], false, bf,
                                                       (short)0, acc2[t], false, false);
    }
  }

  // silu + mask + segment-sum scatter (lane owns column n of 8 rows)
  const int4 ri0 = *(const int4*)(row + e0 + rhalf);
  const int4 ri1 = *(const int4*)(row + e0 + rhalf + 4);
  const v8f mk = *(const v8f*)(emask + e0 + rhalf);
  const int ridx[8] = {ri0.x, ri0.y, ri0.z, ri0.w, ri1.x, ri1.y, ri1.z, ri1.w};
#pragma unroll
  for (int r = 0; r < 8; ++r) {
    const int dst = ridx[r] * HDIM;
    const float msk = mk[r];
#pragma unroll
    for (int t = 0; t < 4; ++t) {
      const int cn = t * 16 + n;
      atomicAdd(&agg[dst + cn], silu_f(acc2[t][r] + b2[cn]) * msk);
    }
  }
}

// ---------------------------------------------------------------- node MLP
// h_new = silu([h|agg|h0] @ W1 + b1) @ W2 + b2   (no silu after GEMM2)
__global__ void __launch_bounds__(128)
k_node(const _Float16* h16, const float* __restrict__ agg,
       const _Float16* __restrict__ h016,
       const _Float16* __restrict__ W1s, const float* __restrict__ b1,
       const _Float16* __restrict__ W2s, const float* __restrict__ b2,
       _Float16* h16out) {
  __shared__ _Float16 lds[4][16 * LSTR];
  const int wave = threadIdx.x >> 5, lane = threadIdx.x & 31;
  const int tile = blockIdx.x * 4 + wave;
  const int m = lane & 15;
  const int kb = (lane >> 4) << 3;
  const int i0 = kb >> 3;
  const int n = lane & 15;
  const int rhalf = (lane >> 4) << 3;
  const v16h* W1f = (const v16h*)W1s;
  const v16h* W2f = (const v16h*)W2s;

  const int nodeA = tile * 16 + m;
  const v8h* hr8 = (const v8h*)(h16 + nodeA * HDIM);
  const float* ar = agg + nodeA * HDIM;
  const v8h* h08 = (const v8h*)(h016 + nodeA * 16);

  v16h a[5];
  a[0] = frag16(hr8[i0],     hr8[i0 + 2]);
  a[1] = frag16(hr8[i0 + 4], hr8[i0 + 6]);
  a[2] = frag16f(*(const v8f*)(ar + kb),      *(const v8f*)(ar + kb + 16));
  a[3] = frag16f(*(const v8f*)(ar + 32 + kb), *(const v8f*)(ar + 32 + kb + 16));
  a[4] = frag16(h08[i0], v8h{});   // h0 cols 11..15 zero-padded; K>=16 is zero

  v8f acc[4] = {};
#pragma unroll
  for (int t = 0; t < 4; ++t) {
#pragma unroll
    for (int c = 0; c < 5; ++c) {
      v16h bf = W1f[(c * 4 + t) * 32 + lane];
      acc[t] = __builtin_amdgcn_wmma_f32_16x16x32_f16(false, a[c], false, bf,
                                                      (short)0, acc[t], false, false);
    }
  }

  _Float16* tl = lds[wave];
#pragma unroll
  for (int t = 0; t < 4; ++t) {
    const int cn = t * 16 + n;
    const float bb = b1[cn];
#pragma unroll
    for (int r = 0; r < 8; ++r)
      tl[(rhalf + r) * LSTR + cn] = (_Float16)silu_f(acc[t][r] + bb);
  }

  const v8h* tl8 = (const v8h*)(tl + m * LSTR);
  v16h a2[2];
  a2[0] = frag16(tl8[i0],     tl8[i0 + 2]);
  a2[1] = frag16(tl8[i0 + 4], tl8[i0 + 6]);

  v8f acc2[4] = {};
#pragma unroll
  for (int t = 0; t < 4; ++t) {
#pragma unroll
    for (int c = 0; c < 2; ++c) {
      v16h bf = W2f[(c * 4 + t) * 32 + lane];
      acc2[t] = __builtin_amdgcn_wmma_f32_16x16x32_f16(false, a2[c], false, bf,
                                                       (short)0, acc2[t], false, false);
    }
  }

#pragma unroll
  for (int r = 0; r < 8; ++r) {
    const int node = tile * 16 + rhalf + r;
#pragma unroll
    for (int t = 0; t < 4; ++t) {
      const int cn = t * 16 + n;
      h16out[node * HDIM + cn] = (_Float16)(acc2[t][r] + b2[cn]);
    }
  }
}

// ---------------------------------------------------------------- node decoder + pool
// hg[g] = sum_{16 nodes} (silu(h @ dW1 + b1) @ dW2 + b2) * node_mask ; tile == graph
__global__ void __launch_bounds__(128)
k_dec(const _Float16* __restrict__ h16,
      const _Float16* __restrict__ W1s, const float* __restrict__ b1,
      const _Float16* __restrict__ W2s, const float* __restrict__ b2,
      const float* __restrict__ nmask, float* __restrict__ hg) {
  __shared__ _Float16 lds[4][16 * LSTR];
  const int wave = threadIdx.x >> 5, lane = threadIdx.x & 31;
  const int tile = blockIdx.x * 4 + wave;
  const int m = lane & 15;
  const int kb = (lane >> 4) << 3;
  const int i0 = kb >> 3;
  const int n = lane & 15;
  const int rhalf = (lane >> 4) << 3;
  const v16h* W1f = (const v16h*)W1s;
  const v16h* W2f = (const v16h*)W2s;

  const v8h* hr8 = (const v8h*)(h16 + (tile * 16 + m) * HDIM);
  v16h a[2];
  a[0] = frag16(hr8[i0],     hr8[i0 + 2]);
  a[1] = frag16(hr8[i0 + 4], hr8[i0 + 6]);

  v8f acc[4] = {};
#pragma unroll
  for (int t = 0; t < 4; ++t) {
#pragma unroll
    for (int c = 0; c < 2; ++c) {
      v16h bf = W1f[(c * 4 + t) * 32 + lane];
      acc[t] = __builtin_amdgcn_wmma_f32_16x16x32_f16(false, a[c], false, bf,
                                                      (short)0, acc[t], false, false);
    }
  }

  _Float16* tl = lds[wave];
#pragma unroll
  for (int t = 0; t < 4; ++t) {
    const int cn = t * 16 + n;
    const float bb = b1[cn];
#pragma unroll
    for (int r = 0; r < 8; ++r)
      tl[(rhalf + r) * LSTR + cn] = (_Float16)silu_f(acc[t][r] + bb);
  }

  const v8h* tl8 = (const v8h*)(tl + m * LSTR);
  v16h a2[2];
  a2[0] = frag16(tl8[i0],     tl8[i0 + 2]);
  a2[1] = frag16(tl8[i0 + 4], tl8[i0 + 6]);

  v8f acc2[4] = {};
#pragma unroll
  for (int t = 0; t < 4; ++t) {
#pragma unroll
    for (int c = 0; c < 2; ++c) {
      v16h bf = W2f[(c * 4 + t) * 32 + lane];
      acc2[t] = __builtin_amdgcn_wmma_f32_16x16x32_f16(false, a2[c], false, bf,
                                                       (short)0, acc2[t], false, false);
    }
  }

  // pool 16 rows of this tile (== one graph): 8 local rows + other lane half
  const v8f msk = *(const v8f*)(nmask + tile * 16 + rhalf);
#pragma unroll
  for (int t = 0; t < 4; ++t) {
    const int cn = t * 16 + n;
    const float b2c = b2[cn];
    float s = 0.f;
#pragma unroll
    for (int r = 0; r < 8; ++r)
      s += (acc2[t][r] + b2c) * msk[r];
    s += __shfl_xor(s, 16, 32);
    if (lane < 16) hg[tile * HDIM + cn] = s;
  }
}

// ---------------------------------------------------------------- graph head
__global__ void k_graph(const float* __restrict__ hg,
                        const float* __restrict__ w1, const float* __restrict__ b1,
                        const float* __restrict__ w2, const float* __restrict__ b2,
                        const int* __restrict__ rxi, const float* __restrict__ rxs,
                        float* __restrict__ pred) {
  __shared__ float t1[HDIM];
  const int g = blockIdx.x;
  const int j = threadIdx.x;
  const float* x = hg + g * HDIM;
  float s = b1[j];
#pragma unroll 8
  for (int k = 0; k < HDIM; ++k) s += x[k] * w1[k * HDIM + j];
  t1[j] = silu_f(s);
  __syncthreads();
  if (j == 0) {
    float o = b2[0];
#pragma unroll 8
    for (int k = 0; k < HDIM; ++k) o += t1[k] * w2[k];
    atomicAdd(&pred[rxi[g]], o * rxs[g]);
  }
}

// ---------------------------------------------------------------- launch
extern "C" void kernel_launch(void* const* d_in, const int* in_sizes, int n_in,
                              void* d_out, int out_size, void* d_ws, size_t ws_size,
                              hipStream_t stream) {
  (void)in_sizes; (void)n_in; (void)out_size; (void)ws_size;
  const float* h0        = (const float*)d_in[0];
  const float* pos       = (const float*)d_in[1];
  const int*   edges     = (const int*)d_in[2];   // [2][E]: row, col
  const float* node_mask = (const float*)d_in[3];
  const float* edge_mask = (const float*)d_in[4];
  const int*   rxi       = (const int*)d_in[5];
  const float* rxs       = (const float*)d_in[6];
  const float* emb_w     = (const float*)d_in[7];
  const float* emb_b     = (const float*)d_in[8];
  const float* edge_w1   = (const float*)d_in[9];
  const float* edge_b1   = (const float*)d_in[10];
  const float* edge_w2   = (const float*)d_in[11];
  const float* edge_b2   = (const float*)d_in[12];
  const float* node_w1   = (const float*)d_in[13];
  const float* node_b1   = (const float*)d_in[14];
  const float* node_w2   = (const float*)d_in[15];
  const float* node_b2   = (const float*)d_in[16];
  const float* dec_w1    = (const float*)d_in[17];
  const float* dec_b1    = (const float*)d_in[18];
  const float* dec_w2    = (const float*)d_in[19];
  const float* dec_b2    = (const float*)d_in[20];
  const float* g_w1      = (const float*)d_in[21];
  const float* g_b1      = (const float*)d_in[22];
  const float* g_w2      = (const float*)d_in[23];
  const float* g_b2      = (const float*)d_in[24];

  float* pred = (float*)d_out;

  char* w = (char*)d_ws;
  auto carve = [&](size_t bytes) -> void* {
    void* p = (void*)w;
    w += (bytes + 255) & ~(size_t)255;
    return p;
  };
  const size_t SZ5 = 5 * 2048;  // swizzled halfs for 5-chunk GEMM (K<=160)
  const size_t SZ2 = 2 * 2048;  // swizzled halfs for 2-chunk GEMM (K=64)
  _Float16* h16  = (_Float16*)carve((size_t)NTOT * HDIM * 2);
  _Float16* h016 = (_Float16*)carve((size_t)NTOT * 16 * 2);
  float*    agg  = (float*)carve((size_t)NTOT * HDIM * 4);
  float*    rad  = (float*)carve((size_t)ETOT * 4);
  float*    hg   = (float*)carve((size_t)GCNT * HDIM * 4);
  _Float16* ew1  = (_Float16*)carve(LAY * SZ5 * 2);
  _Float16* ew2  = (_Float16*)carve(LAY * SZ2 * 2);
  _Float16* nw1  = (_Float16*)carve(LAY * SZ5 * 2);
  _Float16* nw2  = (_Float16*)carve(LAY * SZ2 * 2);
  _Float16* dw1  = (_Float16*)carve(SZ2 * 2);
  _Float16* dw2  = (_Float16*)carve(SZ2 * 2);

  auto swz = [&](const float* W, _Float16* out, int K, int nch) {
    int total = nch * 2048;
    k_swz<<<(total + 255) / 256, 256, 0, stream>>>(W, out, K, nch);
  };
  for (int i = 0; i < LAY; ++i) {
    swz(edge_w1 + (size_t)i * EIN * HDIM,  ew1 + i * SZ5, EIN,  5);
    swz(edge_w2 + (size_t)i * HDIM * HDIM, ew2 + i * SZ2, HDIM, 2);
    swz(node_w1 + (size_t)i * NIN * HDIM,  nw1 + i * SZ5, NIN,  5);
    swz(node_w2 + (size_t)i * HDIM * HDIM, nw2 + i * SZ2, HDIM, 2);
  }
  swz(dec_w1, dw1, HDIM, 2);
  swz(dec_w2, dw2, HDIM, 2);

  k_zero<<<(RCNT + 255) / 256, 256, 0, stream>>>(pred, RCNT);
  k_radial<<<ETOT / 256, 256, 0, stream>>>(pos, edges, edges + ETOT, rad);
  k_embed<<<(NTOT * HDIM) / 256, 256, 0, stream>>>(h0, emb_w, emb_b, h16, h016);

  for (int i = 0; i < LAY; ++i) {
    k_zero<<<(NTOT * HDIM) / 256, 256, 0, stream>>>(agg, NTOT * HDIM);
    k_edge<<<ETOT / 64, 128, 0, stream>>>(h16, rad, edges, edges + ETOT, edge_mask,
                                          ew1 + i * SZ5, edge_b1 + i * HDIM,
                                          ew2 + i * SZ2, edge_b2 + i * HDIM, agg);
    k_node<<<NTOT / 64, 128, 0, stream>>>(h16, agg, h016,
                                          nw1 + i * SZ5, node_b1 + i * HDIM,
                                          nw2 + i * SZ2, node_b2 + i * HDIM, h16);
  }

  k_dec<<<NTOT / 64, 128, 0, stream>>>(h16, dw1, dec_b1, dw2, dec_b2, node_mask, hg);
  k_graph<<<GCNT, HDIM, 0, stream>>>(hg, g_w1, g_b1, g_w2, g_b2, rxi, rxs, pred);
}
